// VideoMemoryCompressorWithExternalWeights_48859547959682
// MI455X (gfx1250) — matrix operand order repaired and checked by reading, not compiled, not run on hardware
//
#include <hip/hip_runtime.h>
#include <hip/hip_bf16.h>
#include <math.h>

// Problem constants (match reference)
#define Bq   8
#define Lq   4096
#define Dq   1024
#define Cq   64
#define Hq   8
#define HDq  128
#define DFFq 4096

typedef __attribute__((ext_vector_type(16))) __bf16 bf16x16;
typedef __attribute__((ext_vector_type(8)))  float  f32x8;

static __device__ __forceinline__ f32x8 wmma_bf16(bf16x16 a, bf16x16 b, f32x8 c) {
  // D = A(16x32 bf16) x B(32x16 bf16) + C(16x16 f32)
  return __builtin_amdgcn_wmma_f32_16x16x32_bf16(false, a, false, b, (short)0, c,
                                                 false, false);
}

// ---- fragment loaders (contiguous k order, same in A and B) -----------
// A 16x32 bf16: lane&15 = row; elements 0..15 -> K = k + (lane<16?0:16)..;
// contiguous.
static __device__ __forceinline__ bf16x16
load_a_f32(const float* A, int lda, int m0, int k, int lane) {
  int row = m0 + (lane & 15);
  int kb  = k + ((lane < 16) ? 0 : 16);
  const float* p = A + (size_t)row * lda + kb;
  bf16x16 a;
#pragma unroll
  for (int i = 0; i < 8; ++i) a[i] = (__bf16)p[i];
#pragma unroll
  for (int i = 0; i < 8; ++i) a[8 + i] = (__bf16)p[8 + i];
  return a;
}

static __device__ __forceinline__ bf16x16
load_a_bf16(const __bf16* A, int lda, int m0, int k, int lane) {
  int row = m0 + (lane & 15);
  int kb  = k + ((lane < 16) ? 0 : 16);
  const __bf16* p = A + (size_t)row * lda + kb;
  bf16x16 a;
#pragma unroll
  for (int i = 0; i < 8; ++i) a[i] = p[i];
#pragma unroll
  for (int i = 0; i < 8; ++i) a[8 + i] = p[8 + i];
  return a;
}

// A-fragment out of an LDS-resident 16 x Dq bf16 tile (m-local = 0..15).
static __device__ __forceinline__ bf16x16
load_a_lds(const __bf16* sA, int k, int lane) {
  int row = lane & 15;
  int kb  = k + ((lane < 16) ? 0 : 16);
  const __bf16* p = sA + row * Dq + kb;
  bf16x16 a;
#pragma unroll
  for (int i = 0; i < 8; ++i) a[i] = p[i];
#pragma unroll
  for (int i = 0; i < 8; ++i) a[8 + i] = p[8 + i];
  return a;
}

// B 32x16 with B[k][n] = W[n][k] (out = x @ W^T): lane&15 = column n,
// 16 contiguous bf16 from W row n starting at k (+16 for lanes >= 16).
static __device__ __forceinline__ bf16x16
load_b_wT(const __bf16* W, int ldw, int n0, int k, int lane) {
  int n  = n0 + (lane & 15);
  int kb = k + ((lane < 16) ? 0 : 16);
  const __bf16* p = W + (size_t)n * ldw + kb;
  bf16x16 b;
#pragma unroll
  for (int i = 0; i < 16; ++i) b[i] = p[i];
  return b;
}

// B 32x16 with B[k][n] = M[k][n] (K-major row-major source, e.g. V).
static __device__ __forceinline__ bf16x16
load_b_rowmajor(const __bf16* M, int ld, int n0, int k, int lane) {
  int n  = n0 + (lane & 15);
  int kb = k + ((lane < 16) ? 0 : 16);
  bf16x16 b;
#pragma unroll
  for (int i = 0; i < 16; ++i) b[i] = M[(size_t)(kb + i) * ld + n];
  return b;
}

// ---- kernels ---------------------------------------------------------------

__global__ void cvt_f32_bf16(const float* __restrict__ src, __bf16* __restrict__ dst,
                             long n) {
  long i = (long)blockIdx.x * blockDim.x + threadIdx.x;
  long stride = (long)gridDim.x * blockDim.x;
  for (; i < n; i += stride) {
    __builtin_prefetch(src + i + stride, 0, 1);   // global_prefetch_b8
    dst[i] = (__bf16)src[i];
  }
}

// Fused K & V projection, async double-buffered through LDS:
//  * 512 threads = 16 waves; waves 0-7 compute K, waves 8-15 compute V
//    (8 accumulators per wave -> no VGPR spills)
//  * each workgroup walks 8 consecutive 16-row Fv slabs; slab s+1 streams
//    into the other LDS buffer via global_load_async_to_lds_b128 while the
//    waves run 8 v_wmma per k-step on slab s (ASYNCcnt pipeline)
//  * Fv is read from HBM exactly once; weights (4 MB bf16) live in L2.
#define KV_SLABS 8
__global__ void __launch_bounds__(512)
kv_proj(const __bf16* __restrict__ FvB,
        const __bf16* __restrict__ Wk, const __bf16* __restrict__ Wv,
        const float* __restrict__ bk, const float* __restrict__ bv,
        __bf16* __restrict__ K, __bf16* __restrict__ V) {
  __shared__ __bf16 sA[2][16 * Dq];                // 2 x 32 KB double buffer
  const int tid    = threadIdx.x;
  const int lane   = tid & 31;
  const int is_v   = tid >> 8;                     // waves 8-15 -> V
  const int n0     = ((tid >> 5) & 7) * 128;       // 8 n-tiles per wave
  const __bf16* Wm = is_v ? Wv : Wk;
  const float*  bm = is_v ? bv : bk;
  __bf16*       Om = is_v ? V  : K;

  auto issue = [&](int buf, int slab) {            // 2048 x 16B chunks, 4/thread
    const char* g =
        (const char*)(FvB + (size_t)(blockIdx.x * KV_SLABS + slab) * 16 * Dq);
    unsigned lbase = (unsigned)(size_t)(&sA[buf][0]);
#pragma unroll
    for (int i = 0; i < 4; ++i) {
      int e = tid + i * 512;
      unsigned    laddr = lbase + e * 16;
      const char* gp    = g + (size_t)e * 16;
      asm volatile("global_load_async_to_lds_b128 %0, %1, off"
                   :: "v"(laddr), "v"(gp) : "memory");
    }
  };

  issue(0, 0);
  for (int s = 0; s < KV_SLABS; ++s) {
    asm volatile("s_wait_asynccnt 0x0" ::: "memory");  // slab s landed
    __syncthreads();                                   // visible to all waves
    if (s + 1 < KV_SLABS) issue((s + 1) & 1, s + 1);   // overlap next slab
    const __bf16* sAc = &sA[s & 1][0];
    int m0 = (blockIdx.x * KV_SLABS + s) * 16;

    f32x8 acc[8];
#pragma unroll
    for (int j = 0; j < 8; ++j) {
      float bval = bm[n0 + j * 16 + (lane & 15)];
#pragma unroll
      for (int i = 0; i < 8; ++i) acc[j][i] = bval;
    }
    for (int k = 0; k < Dq; k += 32) {
      bf16x16 a = load_a_lds(sAc, k, lane);
      bf16x16 b = load_b_wT(Wm, Dq, n0, k, lane);
#pragma unroll
      for (int j = 0; j < 8; ++j) {                // 1-deep B pipeline
        bf16x16 bn = (j < 7) ? load_b_wT(Wm, Dq, n0 + (j + 1) * 16, k, lane) : b;
        acc[j] = wmma_bf16(a, b, acc[j]);
        b = bn;
      }
    }
    int rb = m0 + ((lane < 16) ? 0 : 8);
#pragma unroll
    for (int j = 0; j < 8; ++j) {
      int col = n0 + j * 16 + (lane & 15);
#pragma unroll
      for (int i = 0; i < 8; ++i)
        Om[(size_t)(rb + i) * Dq + col] = (__bf16)acc[j][i];
    }
    // next iteration's wait+barrier fences buffer reuse
  }
}

// out = scale*(A_f32 @ W^T + bias); 8 n-tiles per wave; optional f32/bf16 out.
__global__ void gemm_af32(const float* __restrict__ A, int lda,
                          const __bf16* __restrict__ W, int ldw,
                          const float* __restrict__ bias,
                          float* __restrict__ outF, __bf16* __restrict__ outB,
                          int ldo, int NtGroups, int K, float out_scale) {
  int gid  = blockIdx.x * blockDim.x + threadIdx.x;
  int wave = gid >> 5, lane = gid & 31;
  int mt = wave / NtGroups, ng = wave % NtGroups;
  int m0 = mt * 16, n0 = ng * 128;
  f32x8 acc[8];
#pragma unroll
  for (int j = 0; j < 8; ++j) {
    float bv = bias ? bias[n0 + j * 16 + (lane & 15)] : 0.f;
#pragma unroll
    for (int i = 0; i < 8; ++i) acc[j][i] = bv;
  }
  for (int k = 0; k < K; k += 32) {
    bf16x16 a = load_a_f32(A, lda, m0, k, lane);
    bf16x16 b = load_b_wT(W, ldw, n0, k, lane);
#pragma unroll
    for (int j = 0; j < 8; ++j) {
      bf16x16 bn = (j < 7) ? load_b_wT(W, ldw, n0 + (j + 1) * 16, k, lane) : b;
      acc[j] = wmma_bf16(a, b, acc[j]);
      b = bn;
    }
  }
  int rb = m0 + ((lane < 16) ? 0 : 8);
#pragma unroll
  for (int j = 0; j < 8; ++j) {
    int col = n0 + j * 16 + (lane & 15);
#pragma unroll
    for (int i = 0; i < 8; ++i) {
      float v = acc[j][i] * out_scale;
      if (outF) outF[(size_t)(rb + i) * ldo + col] = v;
      if (outB) outB[(size_t)(rb + i) * ldo + col] = (__bf16)v;
    }
  }
}

// out = relu?(A_bf16 @ W^T + bias); 8 n-tiles per wave.
__global__ void gemm_abf16(const __bf16* __restrict__ A, int lda,
                           const __bf16* __restrict__ W, int ldw,
                           const float* __restrict__ bias,
                           float* __restrict__ outF, __bf16* __restrict__ outB,
                           int ldo, int NtGroups, int K, int relu) {
  int gid  = blockIdx.x * blockDim.x + threadIdx.x;
  int wave = gid >> 5, lane = gid & 31;
  int mt = wave / NtGroups, ng = wave % NtGroups;
  int m0 = mt * 16, n0 = ng * 128;
  f32x8 acc[8];
#pragma unroll
  for (int j = 0; j < 8; ++j) {
    float bv = bias ? bias[n0 + j * 16 + (lane & 15)] : 0.f;
#pragma unroll
    for (int i = 0; i < 8; ++i) acc[j][i] = bv;
  }
  for (int k = 0; k < K; k += 32) {
    bf16x16 a = load_a_bf16(A, lda, m0, k, lane);
    bf16x16 b = load_b_wT(W, ldw, n0, k, lane);
#pragma unroll
    for (int j = 0; j < 8; ++j) {
      bf16x16 bn = (j < 7) ? load_b_wT(W, ldw, n0 + (j + 1) * 16, k, lane) : b;
      acc[j] = wmma_bf16(a, b, acc[j]);
      b = bn;
    }
  }
  int rb = m0 + ((lane < 16) ? 0 : 8);
#pragma unroll
  for (int j = 0; j < 8; ++j) {
    int col = n0 + j * 16 + (lane & 15);
#pragma unroll
    for (int i = 0; i < 8; ++i) {
      float v = acc[j][i];
      if (relu) v = fmaxf(v, 0.f);
      if (outF) outF[(size_t)(rb + i) * ldo + col] = v;
      if (outB) outB[(size_t)(rb + i) * ldo + col] = (__bf16)v;
    }
  }
}

// scores[b,h,c,l] = q_scaled[c,h*HD:] . K[b,l,h*HD:] ; 8 l-tiles per wave.
__global__ void attn_scores(const __bf16* __restrict__ qbf,
                            const __bf16* __restrict__ K,
                            float* __restrict__ sc) {
  int gid  = blockIdx.x * blockDim.x + threadIdx.x;
  int wave = gid >> 5, lane = gid & 31;
  // per (b,h): 4 m-tiles x 32 n-groups = 128 waves
  int bh = wave >> 7, r = wave & 127;
  int b  = bh / Hq, h = bh % Hq;
  int mt = r >> 5, ng = r & 31;
  int m0 = mt * 16, n0 = ng * 128;
  const __bf16* Ab = qbf + h * HDq;
  const __bf16* Kb = K + (size_t)b * Lq * Dq + h * HDq;
  f32x8 acc[8];
#pragma unroll
  for (int j = 0; j < 8; ++j)
#pragma unroll
    for (int i = 0; i < 8; ++i) acc[j][i] = 0.f;
  for (int k = 0; k < HDq; k += 32) {
    bf16x16 a = load_a_bf16(Ab, Dq, m0, k, lane);
    bf16x16 b = load_b_wT(Kb, Dq, n0, k, lane);
#pragma unroll
    for (int j = 0; j < 8; ++j) {
      bf16x16 bn = (j < 7) ? load_b_wT(Kb, Dq, n0 + (j + 1) * 16, k, lane) : b;
      acc[j] = wmma_bf16(a, b, acc[j]);
      b = bn;
    }
  }
  float* O = sc + (size_t)bh * Cq * Lq;
  int rb = m0 + ((lane < 16) ? 0 : 8);
#pragma unroll
  for (int j = 0; j < 8; ++j) {
    int col = n0 + j * 16 + (lane & 15);
#pragma unroll
    for (int i = 0; i < 8; ++i) O[(size_t)(rb + i) * Lq + col] = acc[j][i];
  }
}

// per-batch normalized external weights
__global__ void ew_norm_kernel(const float* __restrict__ ew,
                               const int* __restrict__ mask,
                               float* __restrict__ ewn) {
  int b = blockIdx.x, t = threadIdx.x;
  __shared__ float sh[256];
  float s = 0.f;
  for (int l = t; l < Lq; l += 256)
    s += mask[(size_t)b * Lq + l] ? ew[(size_t)b * Lq + l] : 0.f;
  sh[t] = s; __syncthreads();
  for (int k = 128; k > 0; k >>= 1) { if (t < k) sh[t] += sh[t + k]; __syncthreads(); }
  float inv = 1.f / (sh[0] + 1e-9f);
  __syncthreads();
  for (int l = t; l < Lq; l += 256)
    ewn[(size_t)b * Lq + l] = mask[(size_t)b * Lq + l] ? ew[(size_t)b * Lq + l] * inv : 0.f;
}

// masked softmax + 0.5/0.5 mix with ew + renormalize; in-place on scores
__global__ void softmax_combine(float* __restrict__ sc,
                                const int* __restrict__ mask,
                                const float* __restrict__ ewn) {
  int bhc = blockIdx.x;            // 0 .. B*H*C-1
  int bh  = bhc / Cq;
  int b   = bh / Hq;
  float* row = sc + (size_t)bhc * Lq;
  const int*   mrow = mask + (size_t)b * Lq;
  const float* erow = ewn  + (size_t)b * Lq;
  __shared__ float sh[256];
  int t = threadIdx.x;
  float mx = -3.0e38f;
  for (int l = t; l < Lq; l += 256)
    mx = fmaxf(mx, mrow[l] ? row[l] : -1e9f);
  sh[t] = mx; __syncthreads();
  for (int k = 128; k > 0; k >>= 1) { if (t < k) sh[t] = fmaxf(sh[t], sh[t + k]); __syncthreads(); }
  mx = sh[0]; __syncthreads();
  float sum = 0.f;
  for (int l = t; l < Lq; l += 256) {
    float e = mrow[l] ? __expf(row[l] - mx) : 0.f;
    row[l] = e; sum += e;
  }
  sh[t] = sum; __syncthreads();
  for (int k = 128; k > 0; k >>= 1) { if (t < k) sh[t] += sh[t + k]; __syncthreads(); }
  float inv = 1.f / sh[0]; __syncthreads();
  float rs = 0.f;
  for (int l = t; l < Lq; l += 256) {
    float c = 0.5f * row[l] * inv + 0.5f * erow[l];
    row[l] = c; rs += c;
  }
  sh[t] = rs; __syncthreads();
  for (int k = 128; k > 0; k >>= 1) { if (t < k) sh[t] += sh[t + k]; __syncthreads(); }
  float invR = 1.f / (sh[0] + 1e-9f);
  for (int l = t; l < Lq; l += 256) row[l] *= invR;
}

// combined_single[b,c,l] = mean_h combined[b,h,c,l]
__global__ void comb_single(const float* __restrict__ sc, float* __restrict__ out) {
  long n = (long)Bq * Cq * Lq;
  long i = (long)blockIdx.x * blockDim.x + threadIdx.x;
  long stride = (long)gridDim.x * blockDim.x;
  for (; i < n; i += stride) {
    long b  = i / ((long)Cq * Lq);
    long cl = i % ((long)Cq * Lq);
    float s = 0.f;
#pragma unroll
    for (int h = 0; h < Hq; ++h)
      s += sc[(size_t)(b * Hq + h) * Cq * Lq + cl];
    out[i] = s * (1.f / Hq);
  }
}

// wv[b,c,h*HD+n] = sum_l combined[b,h,c,l] * V[b,l,h*HD+n]
// one wave owns a full 16 x 128 (all 8 HD tiles) strip -> combined read once.
__global__ void attn_wv(const float* __restrict__ comb,
                        const __bf16* __restrict__ V,
                        float* __restrict__ wv) {
  int gid  = blockIdx.x * blockDim.x + threadIdx.x;
  int wave = gid >> 5, lane = gid & 31;
  int bh = wave >> 2, mt = wave & 3;      // 4 m-tiles per (b,h)
  int b  = bh / Hq, h = bh % Hq;
  int m0 = mt * 16;
  const float*  Ab = comb + (size_t)bh * Cq * Lq;
  const __bf16* Vb = V + (size_t)b * Lq * Dq + h * HDq;
  f32x8 acc[8];
#pragma unroll
  for (int j = 0; j < 8; ++j)
#pragma unroll
    for (int i = 0; i < 8; ++i) acc[j][i] = 0.f;
  for (int k = 0; k < Lq; k += 32) {
    bf16x16 a = load_a_f32(Ab, Lq, m0, k, lane);
#pragma unroll
    for (int j = 0; j < 8; ++j)
      acc[j] = wmma_bf16(a, load_b_rowmajor(Vb, Dq, j * 16, k, lane), acc[j]);
  }
  float* O = wv + (size_t)b * Cq * Dq + h * HDq;
  int rb = m0 + ((lane < 16) ? 0 : 8);
#pragma unroll
  for (int j = 0; j < 8; ++j) {
    int col = j * 16 + (lane & 15);
#pragma unroll
    for (int i = 0; i < 8; ++i) O[(size_t)(rb + i) * Dq + col] = acc[j][i];
  }
}

// y = LayerNorm(base + add) * g + beta ; base optionally broadcast over batch
__global__ void ln_residual(const float* __restrict__ base, int broadcastC,
                            const float* __restrict__ add,
                            const float* __restrict__ g, const float* __restrict__ be,
                            float* __restrict__ outF, __bf16* __restrict__ outB) {
  int row = blockIdx.x;                   // 0 .. B*C-1
  int t = threadIdx.x;
  const float* bp = broadcastC ? base + (size_t)(row % Cq) * Dq
                               : base + (size_t)row * Dq;
  const float* ap = add + (size_t)row * Dq;
  __shared__ float sh[256];
  float vals[4];
  float lsum = 0.f;
#pragma unroll
  for (int i = 0; i < 4; ++i) {
    int d = t + i * 256;
    float v = bp[d] + ap[d];
    vals[i] = v; lsum += v;
  }
  sh[t] = lsum; __syncthreads();
  for (int k = 128; k > 0; k >>= 1) { if (t < k) sh[t] += sh[t + k]; __syncthreads(); }
  float mean = sh[0] * (1.f / Dq); __syncthreads();
  float lss = 0.f;
#pragma unroll
  for (int i = 0; i < 4; ++i) { float dv = vals[i] - mean; lss += dv * dv; }
  sh[t] = lss; __syncthreads();
  for (int k = 128; k > 0; k >>= 1) { if (t < k) sh[t] += sh[t + k]; __syncthreads(); }
  float rstd = rsqrtf(sh[0] * (1.f / Dq) + 1e-5f);
#pragma unroll
  for (int i = 0; i < 4; ++i) {
    int d = t + i * 256;
    float y = (vals[i] - mean) * rstd * g[d] + be[d];
    if (outF) outF[(size_t)row * Dq + d] = y;
    if (outB) outB[(size_t)row * Dq + d] = (__bf16)y;
  }
}

__global__ void fill_ones(float* __restrict__ p, int n) {
  int i = blockIdx.x * blockDim.x + threadIdx.x;
  if (i < n) p[i] = 1.0f;
}

// ---- host ------------------------------------------------------------------

extern "C" void kernel_launch(void* const* d_in, const int* in_sizes, int n_in,
                              void* d_out, int out_size, void* d_ws, size_t ws_size,
                              hipStream_t stream) {
  (void)in_sizes; (void)n_in; (void)out_size; (void)ws_size;
  const float* Fv    = (const float*)d_in[0];
  const float* eww   = (const float*)d_in[1];
  const int*   mask  = (const int*)d_in[2];
  const float* query = (const float*)d_in[3];
  const float* Wq = (const float*)d_in[4];  const float* bq = (const float*)d_in[5];
  const float* Wk = (const float*)d_in[6];  const float* bk = (const float*)d_in[7];
  const float* Wv = (const float*)d_in[8];  const float* bv = (const float*)d_in[9];
  const float* Wo = (const float*)d_in[10]; const float* bo = (const float*)d_in[11];
  const float* ln1g = (const float*)d_in[12]; const float* ln1b = (const float*)d_in[13];
  const float* W1 = (const float*)d_in[14]; const float* bf1 = (const float*)d_in[15];
  const float* W2 = (const float*)d_in[16]; const float* bf2 = (const float*)d_in[17];
  const float* ln2g = (const float*)d_in[18]; const float* ln2b = (const float*)d_in[19];

  // outputs: Fv_compress (B,C,D) | compress_mask (B,C) | combined_single (B,C,L)
  float* out_fc   = (float*)d_out;
  float* out_mask = out_fc + (size_t)Bq * Cq * Dq;
  float* out_cs   = out_mask + (size_t)Bq * Cq;

  // workspace carve
  char* w = (char*)d_ws;
  auto carve = [&](size_t bytes) -> char* {
    char* p = w; w += (bytes + 255) & ~(size_t)255; return p;
  };
  __bf16* WqB = (__bf16*)carve((size_t)Dq * Dq * 2);
  __bf16* WkB = (__bf16*)carve((size_t)Dq * Dq * 2);
  __bf16* WvB = (__bf16*)carve((size_t)Dq * Dq * 2);
  __bf16* WoB = (__bf16*)carve((size_t)Dq * Dq * 2);
  __bf16* W1B = (__bf16*)carve((size_t)DFFq * Dq * 2);
  __bf16* W2B = (__bf16*)carve((size_t)Dq * DFFq * 2);
  __bf16* qbf = (__bf16*)carve((size_t)Cq * Dq * 2);
  __bf16* Kbf = (__bf16*)carve((size_t)Bq * Lq * Dq * 2);
  __bf16* Vbf = (__bf16*)carve((size_t)Bq * Lq * Dq * 2);
  float*  sc  = (float*)carve((size_t)Bq * Hq * Cq * Lq * 4);  // 64 MB
  float*  ewn = (float*)carve((size_t)Bq * Lq * 4);
  float*  wvF = (float*)carve((size_t)Bq * Cq * Dq * 4);
  float*  atF = (float*)carve((size_t)Bq * Cq * Dq * 4);
  float*  xF  = (float*)carve((size_t)Bq * Cq * Dq * 4);
  __bf16* xB  = (__bf16*)carve((size_t)Bq * Cq * Dq * 2);
  __bf16* hB  = (__bf16*)carve((size_t)Bq * Cq * DFFq * 2);
  float*  yF  = (float*)carve((size_t)Bq * Cq * Dq * 4);
  // FvB (B*L*D bf16 = 64 MB) aliases the scores buffer: it is consumed by
  // kv_proj before attn_scores first writes sc.
  __bf16* FvB = (__bf16*)sc;

  // 1) weights + Fv -> bf16
  cvt_f32_bf16<<<512, 256, 0, stream>>>(Wq, WqB, (long)Dq * Dq);
  cvt_f32_bf16<<<512, 256, 0, stream>>>(Wk, WkB, (long)Dq * Dq);
  cvt_f32_bf16<<<512, 256, 0, stream>>>(Wv, WvB, (long)Dq * Dq);
  cvt_f32_bf16<<<512, 256, 0, stream>>>(Wo, WoB, (long)Dq * Dq);
  cvt_f32_bf16<<<1024, 256, 0, stream>>>(W1, W1B, (long)DFFq * Dq);
  cvt_f32_bf16<<<1024, 256, 0, stream>>>(W2, W2B, (long)Dq * DFFq);
  cvt_f32_bf16<<<4096, 256, 0, stream>>>(Fv, FvB, (long)Bq * Lq * Dq);

  // 2) q = (query @ Wq^T + bq) * 1/sqrt(HD) -> bf16 [C, D] (batch-invariant)
  gemm_af32<<<4, 256, 0, stream>>>(query, Dq, WqB, Dq, bq,
                                   nullptr, qbf, Dq, /*NtGroups=*/8, Dq,
                                   0.088388347648318447f /* 1/sqrt(128) */);

  // 3) fused K/V projection: 256 workgroups x 512 threads, 8 slabs each,
  //    async double-buffered through LDS
  kv_proj<<<Bq * Lq / 16 / KV_SLABS, 512, 0, stream>>>(FvB, WkB, WvB, bk, bv,
                                                       Kbf, Vbf);

  // 4) normalized external weights
  ew_norm_kernel<<<Bq, 256, 0, stream>>>(eww, mask, ewn);

  // 5) scores = q K^T (scale pre-folded): 8192 waves
  attn_scores<<<1024, 256, 0, stream>>>(qbf, Kbf, sc);

  // 6) masked softmax + mix + renormalize (in-place -> combined)
  softmax_combine<<<Bq * Hq * Cq, 256, 0, stream>>>(sc, mask, ewn);

  // 7) combined_single output (mean over heads)
  comb_single<<<2048, 256, 0, stream>>>(sc, out_cs);

  // 8) wv = combined @ V : 256 waves
  attn_wv<<<32, 256, 0, stream>>>(sc, Vbf, wvF);

  // 9) attn_out = wv @ Wo^T + bo : Mtiles=32 x NtGroups=8 -> 256 waves
  gemm_af32<<<32, 256, 0, stream>>>(wvF, Dq, WoB, Dq, bo,
                                    atF, nullptr, Dq, 8, Dq, 1.0f);

  // 10) x = LN(qry + attn_out)
  ln_residual<<<Bq * Cq, 256, 0, stream>>>(query, 1, atF, ln1g, ln1b, xF, xB);

  // 11) h = relu(x @ W1^T + bf1) : Mtiles=32 x NtGroups=32 -> 1024 waves
  gemm_abf16<<<128, 256, 0, stream>>>(xB, Dq, W1B, Dq, bf1,
                                      nullptr, hB, DFFq, 32, Dq, 1);

  // 12) y = h @ W2^T + bf2 : 256 waves
  gemm_abf16<<<32, 256, 0, stream>>>(hB, DFFq, W2B, DFFq, bf2,
                                     yF, nullptr, Dq, 8, DFFq, 0);

  // 13) Fv_compress = LN(x + y) -> d_out
  ln_residual<<<Bq * Cq, 256, 0, stream>>>(xF, 0, yF, ln2g, ln2b, out_fc, nullptr);

  // 14) compress_mask = ones
  fill_ones<<<(Bq * Cq + 255) / 256, 256, 0, stream>>>(out_mask, Bq * Cq);
}
